// InferenceQuantizer_4303557231310
// MI455X (gfx1250) — matrix operand order repaired and checked by reading, not compiled, not run on hardware
//
#include <hip/hip_runtime.h>

typedef __attribute__((ext_vector_type(16))) __bf16 v16bf;
typedef __attribute__((ext_vector_type(8)))  __bf16 v8bf;
typedef __attribute__((ext_vector_type(8)))  float  v8f;
typedef __attribute__((ext_vector_type(4)))  float  v4f;

#define HIDDEN      4096
#define BS          128
#define NBLK        (HIDDEN / BS)          // 32
#define ROWS_PER_WG 16
#define LDS_STRIDE  (HIDDEN + 8)           // bf16 elements; pad to break bank aliasing
#define THREADS     256
#define NWAVES      8
#define SMEM_BYTES  (ROWS_PER_WG * LDS_STRIDE * 2 + (NWAVES * 16 * 2 + 32) * 4)

// Build partner[] : after the swap, new[:,d] = old[:,partner[d]] and old col n lands at partner[n].
__global__ void build_partner_kernel(const int* __restrict__ perm, int* __restrict__ partner) {
    int j = blockIdx.x * blockDim.x + threadIdx.x;
    if (j < HIDDEN / 2) {
        int a = perm[j];
        int b = perm[HIDDEN / 2 + j];
        partner[a] = b;
        partner[b] = a;
    }
}

__global__ __launch_bounds__(THREADS) void duquant_fused_kernel(
        const float* __restrict__ x, const float* __restrict__ R,
        const int* __restrict__ partner, float* __restrict__ out) {
    extern __shared__ char smem_raw[];
    __bf16* buf   = (__bf16*)smem_raw;                                   // 16 x (4096+8) bf16
    float*  redmn = (float*)(smem_raw + ROWS_PER_WG * LDS_STRIDE * 2);   // [8][16]
    float*  redmx = redmn + NWAVES * 16;                                 // [8][16]
    float*  rowmn = redmx + NWAVES * 16;                                 // [16]
    float*  rowmx = rowmn + 16;                                          // [16]

    const int tid  = threadIdx.x;
    const int wave = tid >> 5;
    const int lane = tid & 31;
    const int l16  = lane & 15;
    const int h    = lane >> 4;          // lane half (WMMA operand layout split)
    const int n0   = wave * 16;          // this wave's N-tile within the 128 block
    const size_t row0 = (size_t)blockIdx.x * ROWS_PER_WG;

    // ---- preload B fragments (R[0], R[1]) in registers, bf16 ----
    // B 32x16 layout: lane half h holds K = h*8 + e (e<8) and 16 + h*8 + (e-8), N = l16.
    v16bf bfr0[4], bfr1[4];
    #pragma unroll
    for (int t = 0; t < 4; ++t) {
        const int kbase = t * 32 + h * 8;
        #pragma unroll
        for (int e = 0; e < 16; ++e) {
            const size_t off = (size_t)(kbase + e + (e >= 8 ? 8 : 0)) * BS + n0 + l16;
            bfr0[t][e] = (__bf16)R[off];
            bfr1[t][e] = (__bf16)R[(size_t)BS * BS + off];
        }
    }

    // ---- rotation 0: global f32 x -> WMMA -> permuted bf16 rows in LDS ----
    // A 16x32 layout: lane half h holds K = h*8 + e (e<8) and 16 + h*8 + (e-8); M = l16.
    const float* xrow = x + (row0 + l16) * (size_t)HIDDEN;
    for (int b = 0; b < NBLK; ++b) {
        v8f c = {0.f, 0.f, 0.f, 0.f, 0.f, 0.f, 0.f, 0.f};
        #pragma unroll
        for (int t = 0; t < 4; ++t) {
            const int k0 = b * BS + t * 32 + h * 8;
            const float4* p0 = (const float4*)(xrow + k0);
            const float4* p1 = (const float4*)(xrow + k0 + 16);
            const float4 fa = p0[0], fb = p0[1], fc = p1[0], fd = p1[1];
            v16bf a;
            a[0]  = (__bf16)fa.x; a[1]  = (__bf16)fa.y; a[2]  = (__bf16)fa.z; a[3]  = (__bf16)fa.w;
            a[4]  = (__bf16)fb.x; a[5]  = (__bf16)fb.y; a[6]  = (__bf16)fb.z; a[7]  = (__bf16)fb.w;
            a[8]  = (__bf16)fc.x; a[9]  = (__bf16)fc.y; a[10] = (__bf16)fc.z; a[11] = (__bf16)fc.w;
            a[12] = (__bf16)fd.x; a[13] = (__bf16)fd.y; a[14] = (__bf16)fd.z; a[15] = (__bf16)fd.w;
            c = __builtin_amdgcn_wmma_f32_16x16x32_bf16(false, a, false, bfr0[t],
                                                        (short)0, c, false, false);
        }
        // C layout: lane owns column N = l16 of this tile, rows M = h*8 + r.
        const int ncol = b * BS + n0 + l16;
        const int dst  = partner[ncol];
        #pragma unroll
        for (int r = 0; r < 8; ++r)
            buf[(size_t)(h * 8 + r) * LDS_STRIDE + dst] = (__bf16)c[r];
    }
    __syncthreads();

    // ---- rotation 1: LDS bf16 -> WMMA -> overwrite block in LDS; track row min/max ----
    float rmin[8], rmax[8];
    #pragma unroll
    for (int r = 0; r < 8; ++r) { rmin[r] = __builtin_inff(); rmax[r] = -__builtin_inff(); }

    for (int b = 0; b < NBLK; ++b) {
        v8f c = {0.f, 0.f, 0.f, 0.f, 0.f, 0.f, 0.f, 0.f};
        #pragma unroll
        for (int t = 0; t < 4; ++t) {
            const int k0 = b * BS + t * 32 + h * 8;
            const __bf16* pr = &buf[(size_t)l16 * LDS_STRIDE + k0];
            const v8bf lo = *(const v8bf*)pr;            // ds_load_b128
            const v8bf hi = *(const v8bf*)(pr + 16);     // ds_load_b128
            const v16bf a = __builtin_shufflevector(lo, hi,
                    0, 1, 2, 3, 4, 5, 6, 7, 8, 9, 10, 11, 12, 13, 14, 15);
            c = __builtin_amdgcn_wmma_f32_16x16x32_bf16(false, a, false, bfr1[t],
                                                        (short)0, c, false, false);
        }
        __syncthreads();   // all waves finished reading block b before it is overwritten
        const int ncol = b * BS + n0 + l16;
        #pragma unroll
        for (int r = 0; r < 8; ++r) {
            const float v = c[r];
            rmin[r] = fminf(rmin[r], v);
            rmax[r] = fmaxf(rmax[r], v);
            buf[(size_t)(h * 8 + r) * LDS_STRIDE + ncol] = (__bf16)v;
        }
    }
    __syncthreads();

    // ---- per-row min/max reduction: shfl within 16-lane halves, then LDS across waves ----
    #pragma unroll
    for (int r = 0; r < 8; ++r) {
        #pragma unroll
        for (int m = 1; m < 16; m <<= 1) {
            rmin[r] = fminf(rmin[r], __shfl_xor(rmin[r], m, 32));
            rmax[r] = fmaxf(rmax[r], __shfl_xor(rmax[r], m, 32));
        }
    }
    if (l16 == 0) {
        #pragma unroll
        for (int r = 0; r < 8; ++r) {
            redmn[wave * 16 + h * 8 + r] = rmin[r];
            redmx[wave * 16 + h * 8 + r] = rmax[r];
        }
    }
    __syncthreads();
    if (tid < 16) {
        float mn = __builtin_inff(), mx = -__builtin_inff();
        #pragma unroll
        for (int w = 0; w < NWAVES; ++w) {
            mn = fminf(mn, redmn[w * 16 + tid]);
            mx = fmaxf(mx, redmx[w * 16 + tid]);
        }
        rowmn[tid] = mn;
        rowmx[tid] = mx;
    }
    __syncthreads();

    // ---- pass 3: fake-quantize, 8 elems/thread: ds_load_b128 -> 2x NT global_store_b128 ----
    for (int vi = tid; vi < ROWS_PER_WG * (HIDDEN / 8); vi += THREADS) {
        const int r    = vi >> 9;              // HIDDEN/8 = 512 vectors per row
        const int col8 = (vi & 511) * 8;
        const float mn = 0.9f * rowmn[r];
        const float mx = 0.9f * rowmx[r];
        const float scale  = fminf(fmaxf((mx - mn) * (1.0f / 15.0f), 1e-5f), 1e4f);
        const float iscale = 1.0f / scale;
        const float zp = rintf(fminf(fmaxf(-mn * iscale, -1e4f), 1e4f));
        const v8bf xv = *(const v8bf*)&buf[(size_t)r * LDS_STRIDE + col8];   // ds_load_b128
        v4f o0, o1;
        #pragma unroll
        for (int e = 0; e < 8; ++e) {
            float q = rintf((float)xv[e] * iscale);
            q = (float)(__bf16)q;                        // reference's bf16 round-trip
            q = fminf(fmaxf(q + zp, 0.0f), 15.0f);
            const float res = (q - zp) * scale;
            if (e < 4) o0[e] = res; else o1[e - 4] = res;
        }
        float* po = out + (row0 + r) * (size_t)HIDDEN + col8;
        __builtin_nontemporal_store(o0, (v4f*)po);       // write-once stream: TH=NT
        __builtin_nontemporal_store(o1, (v4f*)(po + 4));
    }
}

extern "C" void kernel_launch(void* const* d_in, const int* in_sizes, int n_in,
                              void* d_out, int out_size, void* d_ws, size_t ws_size,
                              hipStream_t stream) {
    const float* x    = (const float*)d_in[0];
    const float* R    = (const float*)d_in[1];
    const int*   perm = (const int*)d_in[2];
    float* out = (float*)d_out;
    int* partner = (int*)d_ws;                 // 4096 ints of scratch

    const int rows = in_sizes[0] / HIDDEN;     // 8192
    if (rows <= 0) return;

    build_partner_kernel<<<(HIDDEN / 2 + 255) / 256, 256, 0, stream>>>(perm, partner);

    (void)hipFuncSetAttribute((const void*)duquant_fused_kernel,
                              hipFuncAttributeMaxDynamicSharedMemorySize, SMEM_BYTES);
    duquant_fused_kernel<<<rows / ROWS_PER_WG, THREADS, SMEM_BYTES, stream>>>(
        x, R, partner, out);
}
